// rcda_rebuild_61761629716677
// MI455X (gfx1250) — matrix operand order repaired and checked by reading, not compiled, not run on hardware
//
#include <hip/hip_runtime.h>

// ---------------------------------------------------------------------------
// RCDA (row/col decoupled attention) rebuild for gfx1250 (MI455X, wave32).
// All GEMM-shaped math on v_wmma_f32_16x16x32_bf16 (bf16 in, f32 acc).
// B=4, T=1024, H=W=64, C=256, nh=8, hd=32.
// ---------------------------------------------------------------------------

#define B_  4
#define T_  1024
#define HH  64
#define WW  64
#define C_  256
#define NH  8
#define HD  32
#define SCALING 0.17677669529663687f   // 32^-0.5

typedef __attribute__((ext_vector_type(16))) __bf16 v16bf;
typedef __attribute__((ext_vector_type(8)))  float  v8f;

union Frag { v16bf v; uint4 q[2]; };

__device__ __forceinline__ unsigned short f2bf(float f) {
  unsigned u = __float_as_uint(f);
  u += 0x7FFFu + ((u >> 16) & 1u);       // round-to-nearest-even
  return (unsigned short)(u >> 16);
}

__device__ __forceinline__ v8f wmma_bf16(v16bf a, v16bf b, v8f c) {
  // D = A(16x32) x B(32x16) + C, f32 accumulate
  return __builtin_amdgcn_wmma_f32_16x16x32_bf16(false, a, false, b,
                                                 (short)0, c, false, false);
}

// A-matrix fragment (16x32 bf16) from row-major [row][k], leading dim ld.
// Lane l: row = row0 + (l&15); e0..7 -> k=8*(l>>4)+e ; e8..15 -> k=16+8*(l>>4)+e-8
__device__ __forceinline__ v16bf load_a_frag(const unsigned short* base, int ld,
                                             int row0, int k0) {
  const int l  = threadIdx.x & 31;
  const int hk = (l >> 4) * 8;
  const unsigned short* p = base + (size_t)(row0 + (l & 15)) * ld + k0;
  Frag f;
  f.q[0] = *(const uint4*)(p + hk);
  f.q[1] = *(const uint4*)(p + 16 + hk);
  return f.v;
}

// B-matrix fragment (32x16 bf16) from a B^T row-major buffer [n][k], ld elems.
// Lane l: col = n0 + (l&15); e0..7 -> k=k0+8*(l>>4)+e ; e8..15 -> k=k0+16+8*(l>>4)+e-8
__device__ __forceinline__ v16bf load_b_frag_nk(const unsigned short* base, int ld,
                                                int k0, int n0) {
  const int l = threadIdx.x & 31;
  const unsigned short* p = base + (size_t)(n0 + (l & 15)) * ld + k0 + (l >> 4) * 8;
  Frag f;
  f.q[0] = *(const uint4*)(p);
  f.q[1] = *(const uint4*)(p + 16);
  return f.v;
}

// ---------------------------------------------------------------------------
// small prep kernels
// ---------------------------------------------------------------------------
__global__ void __launch_bounds__(256)
cvt_bf16_kernel(const float* __restrict__ src,
                unsigned short* __restrict__ dst, int n) {
  int i = blockIdx.x * blockDim.x + threadIdx.x;
  int stride = gridDim.x * blockDim.x;
  for (; i < n; i += stride) dst[i] = f2bf(src[i]);
}

// mean over H,W of value -> vmean[B,C]
__global__ void __launch_bounds__(256)
vmean_kernel(const float* __restrict__ value, float* __restrict__ vmean) {
  int b = blockIdx.x, c = threadIdx.x;
  const float* p = value + (size_t)b * HH * WW * C_ + c;
  float s = 0.f;
  for (int i = 0; i < HH * WW; ++i) s += p[(size_t)i * C_];
  vmean[b * C_ + c] = s * (1.0f / (HH * WW));
}

// vavg[b,c] = sigmoid(vmean[b,:] . lin_w[c,:] + lin_b[c])
__global__ void __launch_bounds__(256)
gate_kernel(const float* __restrict__ vmean, const float* __restrict__ lw,
            const float* __restrict__ lb, float* __restrict__ vavg) {
  int b = blockIdx.x, c = threadIdx.x;
  const float* m = vmean + b * C_;
  const float* w = lw + (size_t)c * C_;
  float s = lb[c];
  for (int k = 0; k < C_; ++k) s += m[k] * w[k];
  vavg[b * C_ + c] = 1.f / (1.f + __expf(-s));
}

// mean over one axis of [B,64,64,C] -> bf16 [B,64,C]
__global__ void __launch_bounds__(256)
kmean_kernel(const float* __restrict__ src, unsigned short* __restrict__ dst,
             int keep_stride, int red_stride) {
  int bk = blockIdx.x;           // b*64 + k
  int b = bk >> 6, k = bk & 63;
  int c = threadIdx.x;
  const float* p = src + (size_t)b * HH * WW * C_ + (size_t)k * keep_stride + c;
  float s = 0.f;
  for (int r = 0; r < 64; ++r) s += p[(size_t)r * red_stride];
  dst[(size_t)bk * C_ + c] = f2bf(s * (1.f / 64.f));
}

// ---------------------------------------------------------------------------
// generic WMMA GEMM: Y = epilogue( X[M,256](bf16) @ Wt[256,256](bf16,[n][k]) + bias )
// mode 0: *= SCALING, store bf16 [M,C]
// mode 1: *= vavg[row/64, col], store bf16 [M,C]          (k_row / k_col)
// mode 2: store bf16 transposed vT[b,n][w*32+d][h]        (value proj)
// mode 3: store f32 to d_out at [t, b, c]                 (out proj)
// one wave per 16x16 output tile; K-loop of 8 WMMAs
// ---------------------------------------------------------------------------
__global__ void __launch_bounds__(128)
gemm_wmma_kernel(const unsigned short* __restrict__ X,
                 const unsigned short* __restrict__ Wt,
                 const float* __restrict__ bias,
                 int M, int mode,
                 const float* __restrict__ vavg,
                 unsigned short* __restrict__ Ybf,
                 float* __restrict__ Yf32) {
  const int wave = (blockIdx.x * blockDim.x + threadIdx.x) >> 5;
  const int nt = wave & 15;          // N tile (C_/16 == 16)
  const int mt = wave >> 4;          // M tile
  if (mt * 16 >= M) return;
  const int l = threadIdx.x & 31;

  v8f acc = {};
#pragma unroll
  for (int k0 = 0; k0 < C_; k0 += 32) {
    v16bf a = load_a_frag(X, C_, mt * 16, k0);
    v16bf b = load_b_frag_nk(Wt, C_, k0, nt * 16);
    acc = wmma_bf16(a, b, acc);
  }

  const int col = nt * 16 + (l & 15);
  const float bv = bias[col];
#pragma unroll
  for (int j = 0; j < 8; ++j) {
    const int row = mt * 16 + j + 8 * (l >> 4);
    float y = acc[j] + bv;
    if (mode == 0) {
      Ybf[(size_t)row * C_ + col] = f2bf(y * SCALING);
    } else if (mode == 1) {
      y *= vavg[(row >> 6) * C_ + col];
      Ybf[(size_t)row * C_ + col] = f2bf(y);
    } else if (mode == 2) {
      const int b = row >> 12, h = (row >> 6) & 63, w = row & 63;
      const int n = col >> 5,  d = col & 31;
      Ybf[((size_t)(b * NH + n) * 2048 + w * 32 + d) * 64 + h] = f2bf(y);
    } else {
      const int b = row >> 10, t = row & 1023;
      Yf32[((size_t)t * B_ + b) * C_ + col] = y;
    }
  }
}

// ---------------------------------------------------------------------------
// scores + softmax: per wave one (b, n, t-tile of 16)
// S[16 x 64] = Q[b, t0:t0+16, n*32:+32] @ K[b, 0:64, n*32:+32]^T  (one WMMA/N-tile)
// ---------------------------------------------------------------------------
__global__ void __launch_bounds__(128)
scores_softmax_kernel(const unsigned short* __restrict__ Qbf,
                      const unsigned short* __restrict__ Kbf,
                      float* __restrict__ attn_f32,
                      unsigned short* __restrict__ attn_bf) {
  __shared__ float sS[4][16 * 64];
  const int wave = (blockIdx.x * blockDim.x + threadIdx.x) >> 5;   // 0..2047
  const int wslot = threadIdx.x >> 5;
  const int l = threadIdx.x & 31;
  const int tt = wave & 63;
  const int n  = (wave >> 6) & 7;
  const int b  = wave >> 9;
  const int t0 = tt * 16;

  v16bf a = load_a_frag(Qbf + (size_t)b * T_ * C_, C_, t0, n * HD);
  float* S = sS[wslot];
#pragma unroll
  for (int w0 = 0; w0 < 64; w0 += 16) {
    v8f acc = {};
    v16bf bf = load_b_frag_nk(Kbf + (size_t)b * 64 * C_, C_, n * HD, w0);
    acc = wmma_bf16(a, bf, acc);
#pragma unroll
    for (int j = 0; j < 8; ++j)
      S[(j + 8 * (l >> 4)) * 64 + w0 + (l & 15)] = acc[j];
  }
  __syncthreads();

  // lane owns row (l&15), half (l>>4)*32 .. +31
  const int r  = l & 15;
  const int c0 = (l >> 4) * 32;
  float mx = -1e30f;
#pragma unroll
  for (int i = 0; i < 32; ++i) mx = fmaxf(mx, S[r * 64 + c0 + i]);
  mx = fmaxf(mx, __shfl_xor(mx, 16, 32));
  float e[32], sum = 0.f;
#pragma unroll
  for (int i = 0; i < 32; ++i) { e[i] = __expf(S[r * 64 + c0 + i] - mx); sum += e[i]; }
  sum += __shfl_xor(sum, 16, 32);
  const float inv = 1.f / sum;
  const size_t base = ((size_t)(b * NH + n) * T_ + t0 + r) * 64 + c0;
#pragma unroll
  for (int i = 0; i < 32; ++i) {
    const float p = e[i] * inv;
    attn_f32[base + i] = p;
    if (attn_bf) attn_bf[base + i] = f2bf(p);
  }
}

// ---------------------------------------------------------------------------
// heavy contraction: per wave one (b, n, t-tile of 16)
// M2[16t x (w,d)] = attn_col[16t x 64h] @ vT[(w,d) x 64h]^T   (WMMA, K=64)
// out[t, n*32+d]  = sum_w attn_row[t,w] * M2[t,(w,d)]         (fused epilogue)
// ---------------------------------------------------------------------------
__global__ void __launch_bounds__(128)
outcol_contract_kernel(const unsigned short* __restrict__ ac_bf,
                       const float* __restrict__ ar,
                       const unsigned short* __restrict__ vT,
                       unsigned short* __restrict__ out_bf) {
  __shared__ float sAR[4][16 * 64];
  const int wave = (blockIdx.x * blockDim.x + threadIdx.x) >> 5;  // 0..2047
  const int wslot = threadIdx.x >> 5;
  const int l = threadIdx.x & 31;
  const int tt = wave & 63;
  const int bn = wave >> 6;                 // b*NH + n
  const int n = bn & 7, b = bn >> 3;
  const int t0 = tt * 16;

  const unsigned short* acb = ac_bf + (size_t)bn * T_ * 64;
  v16bf a0 = load_a_frag(acb, 64, t0, 0);
  v16bf a1 = load_a_frag(acb, 64, t0, 32);

  // stage attn_row tile [16][64] in LDS
  const float* arp = ar + ((size_t)bn * T_ + t0) * 64;
  float* SR = sAR[wslot];
  for (int i = l; i < 1024; i += 32) SR[i] = arp[i];
  __syncthreads();

  const unsigned short* vb = vT + (size_t)bn * 2048 * 64;
  float oe[8] = {0,0,0,0,0,0,0,0};
  float oo[8] = {0,0,0,0,0,0,0,0};
  const int tl0 = 8 * (l >> 4);
  for (int w = 0; w < 64; ++w) {
    // prefetch the vT stream 4 w-iterations (16 KB) ahead
    if (w + 4 < 64)
      __builtin_prefetch(vb + (size_t)(w + 4) * 32 * 64 + (l & 15) * 64, 0, 1);
    // hoist the 8 attn_row values for this w (reused by both N-halves)
    float arw[8];
#pragma unroll
    for (int j = 0; j < 8; ++j) arw[j] = SR[(tl0 + j) * 64 + w];
#pragma unroll
    for (int half = 0; half < 2; ++half) {
      const int wd0 = w * 32 + half * 16;
      v16bf b0 = load_b_frag_nk(vb, 64, 0,  wd0);
      v16bf b1 = load_b_frag_nk(vb, 64, 32, wd0);
      v8f acc = {};
      acc = wmma_bf16(a0, b0, acc);
      acc = wmma_bf16(a1, b1, acc);
      float* o = half ? oo : oe;
#pragma unroll
      for (int j = 0; j < 8; ++j) o[j] += arw[j] * acc[j];
    }
  }
  const int d0 = l & 15;
#pragma unroll
  for (int j = 0; j < 8; ++j) {
    const int t = t0 + tl0 + j;
    const size_t base = ((size_t)b * T_ + t) * C_ + n * HD;
    out_bf[base + d0]      = f2bf(oe[j]);
    out_bf[base + 16 + d0] = f2bf(oo[j]);
  }
}

// ---------------------------------------------------------------------------
// grid[t,b,h,w] = (1/nh) * sum_n ar[b,n,t,w] * ac[b,n,t,h]  — K=8 outer products
// one 256-thread block per (b,t); bandwidth-bound (67 MB output)
// ---------------------------------------------------------------------------
__global__ void __launch_bounds__(256)
grid_kernel(const float* __restrict__ ar, const float* __restrict__ ac,
            float* __restrict__ gout) {
  const int b = blockIdx.x >> 10;
  const int t = blockIdx.x & 1023;
  __shared__ float sr[NH][64], sc[NH][64];
  const int tid = threadIdx.x;
  for (int i = tid; i < NH * 64; i += 256) {
    const int n = i >> 6, x = i & 63;
    const size_t base = ((size_t)(b * NH + n) * T_ + t) * 64 + x;
    sr[n][x] = ar[base];
    sc[n][x] = ac[base];
  }
  __syncthreads();
  float* g = gout + ((size_t)t * B_ + b) * (HH * WW);
  for (int i = tid; i < HH * WW; i += 256) {
    const int h = i >> 6, w = i & 63;
    float s = 0.f;
#pragma unroll
    for (int n = 0; n < NH; ++n) s += sc[n][h] * sr[n][w];
    g[i] = s * (1.0f / NH);
  }
}

// ---------------------------------------------------------------------------
extern "C" void kernel_launch(void* const* d_in, const int* in_sizes, int n_in,
                              void* d_out, int out_size, void* d_ws, size_t ws_size,
                              hipStream_t stream) {
  (void)in_sizes; (void)n_in; (void)out_size; (void)ws_size;
  const float* query_row = (const float*)d_in[0];
  const float* query_col = (const float*)d_in[1];
  const float* key_row   = (const float*)d_in[2];
  const float* key_col   = (const float*)d_in[3];
  const float* value     = (const float*)d_in[4];
  const float* ipw       = (const float*)d_in[5];   // [5C, C]
  const float* ipb       = (const float*)d_in[6];   // [5C]
  const float* out_w     = (const float*)d_in[7];
  const float* out_b     = (const float*)d_in[8];
  const float* lin_w     = (const float*)d_in[9];
  const float* lin_b     = (const float*)d_in[10];

  char* wsb = (char*)d_ws;
  size_t off = 0;
  auto alloc = [&](size_t bytes) -> void* {
    void* p = wsb + off;
    off = (off + bytes + 255) & ~(size_t)255;
    return p;
  };
  unsigned short* qrow_in = (unsigned short*)alloc((size_t)B_*T_*C_*2);      // 2MB
  unsigned short* qcol_in = (unsigned short*)alloc((size_t)B_*T_*C_*2);      // 2MB
  unsigned short* val_in  = (unsigned short*)alloc((size_t)B_*HH*WW*C_*2);   // 8MB
  unsigned short* w5_bf   = (unsigned short*)alloc((size_t)5*C_*C_*2);
  unsigned short* outw_bf = (unsigned short*)alloc((size_t)C_*C_*2);
  unsigned short* krm_bf  = (unsigned short*)alloc((size_t)B_*64*C_*2);
  unsigned short* kcm_bf  = (unsigned short*)alloc((size_t)B_*64*C_*2);
  float*          vmean   = (float*)alloc((size_t)B_*C_*4);
  float*          vavg    = (float*)alloc((size_t)B_*C_*4);
  unsigned short* qrow_bf = (unsigned short*)alloc((size_t)B_*T_*C_*2);
  unsigned short* qcol_bf = (unsigned short*)alloc((size_t)B_*T_*C_*2);
  unsigned short* kr_bf   = (unsigned short*)alloc((size_t)B_*64*C_*2);
  unsigned short* kc_bf   = (unsigned short*)alloc((size_t)B_*64*C_*2);
  unsigned short* vT_bf   = (unsigned short*)alloc((size_t)B_*NH*2048*64*2); // 8MB
  float*          ar_f32  = (float*)alloc((size_t)B_*NH*T_*64*4);            // 8MB
  float*          ac_f32  = (float*)alloc((size_t)B_*NH*T_*64*4);            // 8MB
  unsigned short* ac_bf   = (unsigned short*)alloc((size_t)B_*NH*T_*64*2);   // 4MB
  unsigned short* out_bf  = (unsigned short*)alloc((size_t)B_*T_*C_*2);      // 2MB

  float* attn_output = (float*)d_out;                       // [T,B,C]
  float* grid_out    = (float*)d_out + (size_t)T_*B_*C_;    // [T,B,H,W]

  // exact block count for gemm_wmma_kernel: one wave per 16x16 tile, 4 waves/block
  auto gemm_blocks = [](int M) -> int { return ((M / 16) * (C_ / 16) * 32) / 128; };

  // --- prep: f32 -> bf16 conversions ---
  {
    int n;
    n = B_*T_*C_;      cvt_bf16_kernel<<<(n+255)/256, 256, 0, stream>>>(query_row, qrow_in, n);
    n = B_*T_*C_;      cvt_bf16_kernel<<<(n+255)/256, 256, 0, stream>>>(query_col, qcol_in, n);
    n = B_*HH*WW*C_;   cvt_bf16_kernel<<<(n+255)/256, 256, 0, stream>>>(value, val_in, n);
    n = 5*C_*C_;       cvt_bf16_kernel<<<(n+255)/256, 256, 0, stream>>>(ipw, w5_bf, n);
    n = C_*C_;         cvt_bf16_kernel<<<(n+255)/256, 256, 0, stream>>>(out_w, outw_bf, n);
  }

  // --- pooled sigmoid gate ---
  vmean_kernel<<<B_, C_, 0, stream>>>(value, vmean);
  gate_kernel<<<B_, C_, 0, stream>>>(vmean, lin_w, lin_b, vavg);

  // --- mean-before-projection (linearity): kr_mean[b,w,c], kc_mean[b,h,c] ---
  kmean_kernel<<<B_*64, C_, 0, stream>>>(key_row, krm_bf, /*keep=w*/C_, /*red=h*/WW*C_);
  kmean_kernel<<<B_*64, C_, 0, stream>>>(key_col, kcm_bf, /*keep=h*/WW*C_, /*red=w*/C_);

  // --- projections (WMMA) ---
  gemm_wmma_kernel<<<gemm_blocks(B_*T_), 128, 0, stream>>>(
      qrow_in, w5_bf + 0*C_*C_, ipb + 0*C_, B_*T_, 0, nullptr, qrow_bf, nullptr);
  gemm_wmma_kernel<<<gemm_blocks(B_*T_), 128, 0, stream>>>(
      qcol_in, w5_bf + 1*C_*C_, ipb + 1*C_, B_*T_, 0, nullptr, qcol_bf, nullptr);
  gemm_wmma_kernel<<<gemm_blocks(B_*64), 128, 0, stream>>>(
      krm_bf, w5_bf + 2*C_*C_, ipb + 2*C_, B_*64, 1, vavg, kr_bf, nullptr);
  gemm_wmma_kernel<<<gemm_blocks(B_*64), 128, 0, stream>>>(
      kcm_bf, w5_bf + 3*C_*C_, ipb + 3*C_, B_*64, 1, vavg, kc_bf, nullptr);
  gemm_wmma_kernel<<<gemm_blocks(B_*HH*WW), 128, 0, stream>>>(
      val_in, w5_bf + 4*C_*C_, ipb + 4*C_, B_*HH*WW, 2, nullptr, vT_bf, nullptr);

  // --- attention scores + softmax (WMMA, K=hd=32) ---
  scores_softmax_kernel<<<2048*32/128, 128, 0, stream>>>(qrow_bf, kr_bf, ar_f32, nullptr);
  scores_softmax_kernel<<<2048*32/128, 128, 0, stream>>>(qcol_bf, kc_bf, ac_f32, ac_bf);

  // --- heavy contraction: attn_col @ V (WMMA, K=H=64) fused with row combine ---
  outcol_contract_kernel<<<2048*32/128, 128, 0, stream>>>(ac_bf, ar_f32, vT_bf, out_bf);

  // --- output projection, write f32 to d_out as [T,B,C] ---
  gemm_wmma_kernel<<<gemm_blocks(B_*T_), 128, 0, stream>>>(
      out_bf, outw_bf, out_b, B_*T_, 3, nullptr, nullptr, attn_output);

  // --- grid map [T,B,H,W] (bandwidth-bound 67 MB write) ---
  grid_kernel<<<B_*T_, 256, 0, stream>>>(ar_f32, ac_f32, grid_out);
}
